// LSTM_attention_83141976916056
// MI455X (gfx1250) — matrix operand order repaired and checked by reading, not compiled, not run on hardware
//
#include <hip/hip_runtime.h>
#include <math.h>

// ---------------- problem constants ----------------
#define LAYERS   3
#define T_STEPS  38
#define T_PAD    48            // T padded to multiple of 16 for the [v,t] GEMM
#define Hdim     512
#define Vdim     10000
#define Bdim     256
#define L_ENC    49
#define G3H      (3 * Hdim)    // 1536
#define WPB      8             // waves per 256-thread block (wave32)

typedef __bf16 bf16;
typedef bf16  v16bf __attribute__((ext_vector_type(16)));
typedef float v8f   __attribute__((ext_vector_type(8)));

// ---------------- WMMA helpers (CDNA5, wave32) ----------------
__device__ __forceinline__ v8f wmma_bf16(v16bf a, v16bf b, v8f c) {
  return __builtin_amdgcn_wmma_f32_16x16x32_bf16(
      false, a, false, b, (short)0, c, false, false);
}

// A fragment (16x32 bf16) from an fp32 row (converted on the fly).
// ISA 7.12.2: lanes 0-15 hold K {0..7},{16..23}; lanes 16-31 hold K {8..15},{24..31}.
__device__ __forceinline__ v16bf load_a_row(const float* __restrict__ rowp,
                                            int k0, int lane) {
  const int koff = ((lane >> 4) & 1) * 8;
  const float* p = rowp + k0 + koff;
  v16bf a;
#pragma unroll
  for (int j = 0; j < 8; ++j) a[j]     = (bf16)p[j];
#pragma unroll
  for (int j = 0; j < 8; ++j) a[j + 8] = (bf16)p[16 + j];
  return a;
}

// A fragment directly from bf16 row (no conversion).
__device__ __forceinline__ v16bf load_a_row_bf(const bf16* __restrict__ rowp,
                                               int k0, int lane) {
  const int koff = ((lane >> 4) & 1) * 8;
  const bf16* p = rowp + k0 + koff;
  v16bf a;
#pragma unroll
  for (int j = 0; j < 8; ++j) a[j]     = p[j];
#pragma unroll
  for (int j = 0; j < 8; ++j) a[j + 8] = p[16 + j];
  return a;
}

// B fragment (32x16): column n per lane&15; lanes 0-15 K {0..7},{16..23}; lanes 16-31 K {8..15},{24..31}.
__device__ __forceinline__ v16bf load_b_frag(const bf16* __restrict__ W,
                                             int ldw, int n, int k0, int lane) {
  const int koff = ((lane >> 4) & 1) * 8;
  const bf16* p = W + (size_t)n * ldw + k0 + koff;
  v16bf b;
#pragma unroll
  for (int j = 0; j < 8; ++j) b[j]     = p[j];
#pragma unroll
  for (int j = 0; j < 8; ++j) b[j + 8] = p[16 + j];
  return b;
}

// ---- generic GEMM, fp32 A:  C[M,N] = A * W^T + bias ; NBLK n-tiles per wave ----
template <int NBLK>
__global__ void gemm_f32a_nb(const float* __restrict__ A, int lda,
                             const bf16* __restrict__ W, int ldw,
                             const float* __restrict__ bias,
                             float* __restrict__ C, int ldc,
                             int M, int N, int K) {
  const int lane = threadIdx.x & 31;
  const int wave = threadIdx.x >> 5;
  const int ntg  = (N >> 4) / NBLK;
  const int tid  = blockIdx.x * WPB + wave;
  if (tid >= (M >> 4) * ntg) return;              // wave-uniform exit
  const int mt = tid / ntg, ng = tid % ntg;
  const int n0 = ng * NBLK * 16 + (lane & 15);
  const float* rowp = A + (size_t)((mt << 4) + (lane & 15)) * lda;

  v8f acc[NBLK] = {};
  for (int k0 = 0; k0 < K; k0 += 32) {
    v16bf a = load_a_row(rowp, k0, lane);
#pragma unroll
    for (int q = 0; q < NBLK; ++q) {
      v16bf b = load_b_frag(W, ldw, n0 + q * 16, k0, lane);
      acc[q] = wmma_bf16(a, b, acc[q]);
    }
  }
  const int mb = (mt << 4) + ((lane >> 4) ? 8 : 0);
#pragma unroll
  for (int q = 0; q < NBLK; ++q) {
    const int n = n0 + q * 16;
    const float bv = bias ? bias[n] : 0.0f;
#pragma unroll
    for (int r = 0; r < 8; ++r)
      C[(size_t)(mb + r) * ldc + n] = acc[q][r] + bv;
  }
}

// ---- generic GEMM, bf16 A (no conversion on either side) ----
template <int NBLK>
__global__ void gemm_bfa_nb(const bf16* __restrict__ A, int lda,
                            const bf16* __restrict__ W, int ldw,
                            const float* __restrict__ bias,
                            float* __restrict__ C, int ldc,
                            int M, int N, int K) {
  const int lane = threadIdx.x & 31;
  const int wave = threadIdx.x >> 5;
  const int ntg  = (N >> 4) / NBLK;
  const int tid  = blockIdx.x * WPB + wave;
  if (tid >= (M >> 4) * ntg) return;
  const int mt = tid / ntg, ng = tid % ntg;
  const int n0 = ng * NBLK * 16 + (lane & 15);
  const bf16* rowp = A + (size_t)((mt << 4) + (lane & 15)) * lda;

  v8f acc[NBLK] = {};
  for (int k0 = 0; k0 < K; k0 += 32) {
    v16bf a = load_a_row_bf(rowp, k0, lane);
#pragma unroll
    for (int q = 0; q < NBLK; ++q) {
      v16bf b = load_b_frag(W, ldw, n0 + q * 16, k0, lane);
      acc[q] = wmma_bf16(a, b, acc[q]);
    }
  }
  const int mb = (mt << 4) + ((lane >> 4) ? 8 : 0);
#pragma unroll
  for (int q = 0; q < NBLK; ++q) {
    const int n = n0 + q * 16;
    const float bv = bias ? bias[n] : 0.0f;
#pragma unroll
    for (int r = 0; r < 8; ++r)
      C[(size_t)(mb + r) * ldc + n] = acc[q][r] + bv;
  }
}

// ---- fused GRU gate GEMMs with 4-way N blocking: gi = x*Wih^T+bih ; gh = h*Whh^T+bhh ----
__global__ void gru_gates_kernel(const float* __restrict__ x,
                                 const float* __restrict__ h,
                                 const bf16* __restrict__ Wih,
                                 const bf16* __restrict__ Whh,
                                 const float* __restrict__ bih,
                                 const float* __restrict__ bhh,
                                 float* __restrict__ gi,
                                 float* __restrict__ gh) {
  const int NBLK = 4;
  const int NTG  = (G3H >> 4) / NBLK;           // 24
  const int lane = threadIdx.x & 31;
  const int wave = threadIdx.x >> 5;
  const int tid  = blockIdx.x * WPB + wave;
  if (tid >= (Bdim >> 4) * NTG) return;
  const int mt = tid / NTG, ng = tid % NTG;
  const int n0 = ng * NBLK * 16 + (lane & 15);
  const float* xr = x + (size_t)((mt << 4) + (lane & 15)) * Hdim;
  const float* hr = h + (size_t)((mt << 4) + (lane & 15)) * Hdim;

  v8f ai[NBLK] = {}, ah[NBLK] = {};
  for (int k0 = 0; k0 < Hdim; k0 += 32) {
    v16bf ax = load_a_row(xr, k0, lane);
    v16bf av = load_a_row(hr, k0, lane);
#pragma unroll
    for (int q = 0; q < NBLK; ++q) {
      v16bf bi = load_b_frag(Wih, Hdim, n0 + q * 16, k0, lane);
      ai[q] = wmma_bf16(ax, bi, ai[q]);
      v16bf bh = load_b_frag(Whh, Hdim, n0 + q * 16, k0, lane);
      ah[q] = wmma_bf16(av, bh, ah[q]);
    }
  }
  const int mb = (mt << 4) + ((lane >> 4) ? 8 : 0);
#pragma unroll
  for (int q = 0; q < NBLK; ++q) {
    const int n = n0 + q * 16;
    const float bvi = bih[n], bvh = bhh[n];
#pragma unroll
    for (int r = 0; r < 8; ++r) {
      gi[(size_t)(mb + r) * G3H + n] = ai[q][r] + bvi;
      gh[(size_t)(mb + r) * G3H + n] = ah[q][r] + bvh;
    }
  }
}

// ---- GRU elementwise gate combine; also maintains bf16 copy of decoder output ----
__global__ void gru_combine_kernel(const float* __restrict__ gi,
                                   const float* __restrict__ gh,
                                   float* __restrict__ h,
                                   float* __restrict__ x,
                                   bf16* __restrict__ predbf_out) {
  const int i = blockIdx.x * blockDim.x + threadIdx.x;
  if (i >= Bdim * Hdim) return;
  const int b = i >> 9, j = i & (Hdim - 1);
  const float* gib = gi + (size_t)b * G3H;
  const float* ghb = gh + (size_t)b * G3H;
  const float r = 1.0f / (1.0f + expf(-(gib[j] + ghb[j])));
  const float z = 1.0f / (1.0f + expf(-(gib[Hdim + j] + ghb[Hdim + j])));
  const float n = tanhf(gib[2 * Hdim + j] + r * ghb[2 * Hdim + j]);
  const float out = (1.0f - z) * n + z * h[i];
  h[i] = out;
  x[i] = out;
  if (predbf_out) predbf_out[i] = (bf16)out;
}

// ---- argmax over vocab + embedding gather ----
__global__ void argmax_embed_kernel(const float* __restrict__ logits,
                                    const float* __restrict__ embed,
                                    float* __restrict__ x) {
  __shared__ float smax[256];
  __shared__ int   sidx[256];
  const int b = blockIdx.x, t = threadIdx.x;
  float best = -INFINITY; int bi = 0;
  for (int v = t; v < Vdim; v += 256) {
    const float val = logits[(size_t)b * Vdim + v];
    if (val > best) { best = val; bi = v; }
  }
  smax[t] = best; sidx[t] = bi;
  __syncthreads();
  for (int s = 128; s > 0; s >>= 1) {
    if (t < s) {
      if (smax[t + s] > smax[t] ||
          (smax[t + s] == smax[t] && sidx[t + s] < sidx[t])) {
        smax[t] = smax[t + s]; sidx[t] = sidx[t + s];
      }
    }
    __syncthreads();
  }
  const int idx = sidx[0];
  for (int j = t; j < Hdim; j += 256)
    x[(size_t)b * Hdim + j] = embed[(size_t)idx * Hdim + j];
}

// ---- attention pieces ----
__global__ void havg_kernel(const float* __restrict__ h, float* __restrict__ havg) {
  const int i = blockIdx.x * blockDim.x + threadIdx.x;
  if (i >= Bdim * Hdim) return;
  havg[i] = (h[i] + h[Bdim * Hdim + i] + h[2 * Bdim * Hdim + i]) * (1.0f / 3.0f);
}

__global__ void attn_kernel(const float* __restrict__ enc3,   // [B,3,H]
                            const float* __restrict__ hid,    // [B,H]
                            const float* __restrict__ oW,     // [H]
                            const float* __restrict__ ob,     // [1]
                            const float* __restrict__ feat,   // [B,L,H]
                            float* __restrict__ hatt) {       // [B,H]
  __shared__ float red[256];
  __shared__ float w[LAYERS];
  const int b = blockIdx.x, t = threadIdx.x;
  float e[LAYERS];
  for (int l = 0; l < LAYERS; ++l) {
    float p = 0.0f;
    for (int j = t; j < Hdim; j += 256)
      p += tanhf(enc3[(size_t)(b * LAYERS + l) * Hdim + j] + hid[(size_t)b * Hdim + j]) * oW[j];
    red[t] = p; __syncthreads();
    for (int s = 128; s > 0; s >>= 1) { if (t < s) red[t] += red[t + s]; __syncthreads(); }
    e[l] = red[0] + ob[0];
    __syncthreads();
  }
  if (t == 0) {
    const float m = fmaxf(e[0], fmaxf(e[1], e[2]));
    const float s0 = expf(e[0] - m), s1 = expf(e[1] - m), s2 = expf(e[2] - m);
    const float inv = 1.0f / (s0 + s1 + s2);
    w[0] = s0 * inv; w[1] = s1 * inv; w[2] = s2 * inv;
  }
  __syncthreads();
  for (int j = t; j < Hdim; j += 256) {
    float acc = 0.0f;
    for (int l = 0; l < LAYERS; ++l)
      acc += w[l] * feat[((size_t)b * L_ENC + l) * Hdim + j];
    hatt[(size_t)b * Hdim + j] = acc;
  }
}

// ---- h_out = concat(h, h_att) @ lin_W^T + lin_b  (M=3B, K=2H, N=H), 4-way N blocking ----
__global__ void lin_kernel(const float* __restrict__ h,     // [3,B,H]
                           const float* __restrict__ hatt,  // [B,H]
                           const bf16* __restrict__ Wlin,   // [H,2H] bf16
                           const float* __restrict__ blin,  // [H]
                           float* __restrict__ hout) {      // [3,B,H]
  const int NBLK = 4;
  const int NTG  = (Hdim >> 4) / NBLK;          // 8
  const int M    = LAYERS * Bdim;
  const int lane = threadIdx.x & 31;
  const int wave = threadIdx.x >> 5;
  const int tid  = blockIdx.x * WPB + wave;
  if (tid >= (M >> 4) * NTG) return;
  const int mt = tid / NTG, ng = tid % NTG;
  const int n0 = ng * NBLK * 16 + (lane & 15);
  const int m  = (mt << 4) + (lane & 15);
  const float* hrow = h    + (size_t)m * Hdim;
  const float* arow = hatt + (size_t)(m & (Bdim - 1)) * Hdim;   // m % B

  v8f acc[NBLK] = {};
  for (int k0 = 0; k0 < 2 * Hdim; k0 += 32) {
    v16bf a = (k0 < Hdim) ? load_a_row(hrow, k0, lane)
                          : load_a_row(arow, k0 - Hdim, lane);
#pragma unroll
    for (int q = 0; q < NBLK; ++q) {
      v16bf bfr = load_b_frag(Wlin, 2 * Hdim, n0 + q * 16, k0, lane);
      acc[q] = wmma_bf16(a, bfr, acc[q]);
    }
  }
  const int mb = (mt << 4) + ((lane >> 4) ? 8 : 0);
#pragma unroll
  for (int q = 0; q < NBLK; ++q) {
    const int n = n0 + q * 16;
    const float bv = blin[n];
#pragma unroll
    for (int r = 0; r < 8; ++r)
      hout[(size_t)(mb + r) * Hdim + n] = acc[q][r] + bv;
  }
}

// ---- final projection, all-bf16, [v,t]-tiled for coalesced T-major stores ----
// out[b,v,t] = proj_W[v,:] . pred[t,b,:] + proj_b[v]
__global__ void final_proj_kernel(const bf16* __restrict__ Wp,      // [V,H] bf16
                                  const bf16* __restrict__ predbf,  // [T_PAD,B,H] bf16
                                  const float* __restrict__ pb,     // [V]
                                  float* __restrict__ out) {        // [B,V,T]
  const int MBLK = 5;
  const int NT   = T_PAD >> 4;                   // 3 column tiles of t
  const int MG   = (Vdim >> 4) / MBLK;           // 125
  const int lane = threadIdx.x & 31;
  const int wave = threadIdx.x >> 5;
  const int tid  = blockIdx.x * WPB + wave;
  if (tid >= MG * NT) return;
  const int b  = blockIdx.y;
  const int mg = tid / NT, tt = tid % NT;
  const int tcol = (tt << 4) + (lane & 15);      // output column t (may be >= 38)
  const bf16* Bbase = predbf + (size_t)b * Hdim; // row t lives at + t*B*H

  v8f acc[MBLK] = {};
  for (int k0 = 0; k0 < Hdim; k0 += 32) {
    v16bf bfr = load_b_frag(Bbase, Bdim * Hdim, tcol, k0, lane);
#pragma unroll
    for (int q = 0; q < MBLK; ++q) {
      const bf16* arow = Wp + (size_t)((mg * MBLK + q) * 16 + (lane & 15)) * Hdim;
      acc[q] = wmma_bf16(load_a_row_bf(arow, k0, lane), bfr, acc[q]);
    }
  }
  const int rsel = (lane >> 4) ? 8 : 0;
  if (tcol < T_STEPS) {                          // mask only the padded stores
#pragma unroll
    for (int q = 0; q < MBLK; ++q) {
      const int vbase = (mg * MBLK + q) * 16 + rsel;
#pragma unroll
      for (int r = 0; r < 8; ++r)
        out[(size_t)b * Vdim * T_STEPS + (size_t)(vbase + r) * T_STEPS + tcol] =
            acc[q][r] + pb[vbase + r];
    }
  }
}

// ---- setup kernels ----
__global__ void cvt_bf16_kernel(const float* __restrict__ in, bf16* __restrict__ out, int n) {
  const int i = blockIdx.x * blockDim.x + threadIdx.x;
  if (i < n) out[i] = (bf16)in[i];
}

__global__ void gather_feat3_kernel(const float* __restrict__ feat, float* __restrict__ feat3) {
  const int i = blockIdx.x * blockDim.x + threadIdx.x;           // B*3*H
  if (i >= Bdim * LAYERS * Hdim) return;
  const int j = i % Hdim;
  const int l = (i / Hdim) % LAYERS;
  const int b = i / (Hdim * LAYERS);
  feat3[i] = feat[((size_t)b * L_ENC + l) * Hdim + j];
}

__global__ void init_kernel(const float* __restrict__ feat, const float* __restrict__ embed,
                            float* __restrict__ h, float* __restrict__ x,
                            bf16* __restrict__ predbf) {
  const int i = blockIdx.x * blockDim.x + threadIdx.x;           // B*H
  if (i >= Bdim * Hdim) return;
  const int b = i >> 9, j = i & (Hdim - 1);
  float s = 0.0f;
  for (int l = 0; l < L_ENC; ++l) s += feat[((size_t)b * L_ENC + l) * Hdim + j];
  s *= (1.0f / L_ENC);
  h[i] = s; h[Bdim * Hdim + i] = s; h[2 * Bdim * Hdim + i] = s;
  const float e0 = embed[j];                                     // <SOS> row 0
  x[i] = e0;
  predbf[i] = (bf16)e0;                                          // pred row 0
  (void)b;
}

// ---------------- host driver ----------------
extern "C" void kernel_launch(void* const* d_in, const int* in_sizes, int n_in,
                              void* d_out, int out_size, void* d_ws, size_t ws_size,
                              hipStream_t stream) {
  (void)in_sizes; (void)n_in; (void)out_size; (void)ws_size;
  const float* feat   = (const float*)d_in[0];
  const float* embed  = (const float*)d_in[1];
  const float* gWih   = (const float*)d_in[2];
  const float* gWhh   = (const float*)d_in[3];
  const float* gbih   = (const float*)d_in[4];
  const float* gbhh   = (const float*)d_in[5];
  const float* projW  = (const float*)d_in[6];
  const float* projb  = (const float*)d_in[7];
  const float* ahW    = (const float*)d_in[8];
  const float* ahb    = (const float*)d_in[9];
  const float* aeW    = (const float*)d_in[10];
  const float* aeb    = (const float*)d_in[11];
  const float* aoW    = (const float*)d_in[12];
  const float* aob    = (const float*)d_in[13];
  const float* linW   = (const float*)d_in[14];
  const float* linb   = (const float*)d_in[15];
  float* out = (float*)d_out;

  // ---- workspace carve ----
  char* ws = (char*)d_ws;
  auto carve = [&](size_t bytes) {
    void* p = (void*)ws;
    ws += (bytes + 255) & ~(size_t)255;
    return p;
  };
  bf16* Wih_bf   = (bf16*)carve((size_t)LAYERS * G3H * Hdim * sizeof(bf16));
  bf16* Whh_bf   = (bf16*)carve((size_t)LAYERS * G3H * Hdim * sizeof(bf16));
  bf16* projW_bf = (bf16*)carve((size_t)Vdim * Hdim * sizeof(bf16));
  bf16* ahW_bf   = (bf16*)carve((size_t)Hdim * Hdim * sizeof(bf16));
  bf16* aeW_bf   = (bf16*)carve((size_t)Hdim * Hdim * sizeof(bf16));
  bf16* linW_bf  = (bf16*)carve((size_t)Hdim * 2 * Hdim * sizeof(bf16));
  bf16* pred_bf  = (bf16*)carve((size_t)T_PAD * Bdim * Hdim * sizeof(bf16)); // padded rows
  float* feat3   = (float*)carve((size_t)Bdim * LAYERS * Hdim * sizeof(float));
  float* enc3    = (float*)carve((size_t)Bdim * LAYERS * Hdim * sizeof(float));
  float* hA      = (float*)carve((size_t)LAYERS * Bdim * Hdim * sizeof(float));
  float* hB      = (float*)carve((size_t)LAYERS * Bdim * Hdim * sizeof(float));
  float* xbuf    = (float*)carve((size_t)Bdim * Hdim * sizeof(float));
  float* gi      = (float*)carve((size_t)Bdim * G3H * sizeof(float));
  float* gh      = (float*)carve((size_t)Bdim * G3H * sizeof(float));
  float* logits  = (float*)carve((size_t)Bdim * Vdim * sizeof(float));
  float* havg    = (float*)carve((size_t)Bdim * Hdim * sizeof(float));
  float* hid     = (float*)carve((size_t)Bdim * Hdim * sizeof(float));
  float* hatt    = (float*)carve((size_t)Bdim * Hdim * sizeof(float));

  const int BH = Bdim * Hdim;
  auto cdiv = [](int a, int b) { return (a + b - 1) / b; };

  // ---- setup: weight conversion fp32 -> bf16 ----
  {
    int n;
    n = LAYERS * G3H * Hdim;
    cvt_bf16_kernel<<<cdiv(n, 256), 256, 0, stream>>>(gWih, Wih_bf, n);
    cvt_bf16_kernel<<<cdiv(n, 256), 256, 0, stream>>>(gWhh, Whh_bf, n);
    n = Vdim * Hdim;
    cvt_bf16_kernel<<<cdiv(n, 256), 256, 0, stream>>>(projW, projW_bf, n);
    n = Hdim * Hdim;
    cvt_bf16_kernel<<<cdiv(n, 256), 256, 0, stream>>>(ahW, ahW_bf, n);
    cvt_bf16_kernel<<<cdiv(n, 256), 256, 0, stream>>>(aeW, aeW_bf, n);
    n = Hdim * 2 * Hdim;
    cvt_bf16_kernel<<<cdiv(n, 256), 256, 0, stream>>>(linW, linW_bf, n);
  }
  // ---- setup: enc_attn over first 3 positions, initial states ----
  gather_feat3_kernel<<<cdiv(Bdim * LAYERS * Hdim, 256), 256, 0, stream>>>(feat, feat3);
  {
    const int tiles = ((Bdim * LAYERS) >> 4) * ((Hdim >> 4) / 4);   // NBLK=4
    gemm_f32a_nb<4><<<cdiv(tiles, WPB), 256, 0, stream>>>(
        feat3, Hdim, aeW_bf, Hdim, aeb, enc3, Hdim,
        Bdim * LAYERS, Hdim, Hdim);
  }
  init_kernel<<<cdiv(BH, 256), 256, 0, stream>>>(feat, embed, hA, xbuf, pred_bf);

  // ---- decode loop: 37 sequential steps ----
  float* hcur = hA;
  float* hnext = hB;
  const int g_gates = cdiv((Bdim >> 4) * ((G3H >> 4) / 4), WPB);     // 384 waves
  const int g_proj  = cdiv((Bdim >> 4) * ((Vdim >> 4) / 5), WPB);    // 2000 waves
  const int g_hid   = cdiv((Bdim >> 4) * ((Hdim >> 4) / 4), WPB);    // 128 waves
  const int g_lin   = cdiv(((LAYERS * Bdim) >> 4) * ((Hdim >> 4) / 4), WPB);

  for (int t = 0; t < T_STEPS - 1; ++t) {
    bf16* predbf_t1 = pred_bf + (size_t)(t + 1) * BH;
    for (int layer = 0; layer < LAYERS; ++layer) {
      gru_gates_kernel<<<g_gates, 256, 0, stream>>>(
          xbuf, hcur + (size_t)layer * BH,
          Wih_bf + (size_t)layer * G3H * Hdim,
          Whh_bf + (size_t)layer * G3H * Hdim,
          gbih + (size_t)layer * G3H, gbhh + (size_t)layer * G3H,
          gi, gh);
      gru_combine_kernel<<<cdiv(BH, 256), 256, 0, stream>>>(
          gi, gh, hcur + (size_t)layer * BH, xbuf,
          (layer == LAYERS - 1) ? predbf_t1 : (bf16*)nullptr);
    }
    // greedy feedback: logits (all-bf16 GEMM) -> argmax -> re-embed into xbuf
    gemm_bfa_nb<5><<<g_proj, 256, 0, stream>>>(
        predbf_t1, Hdim, projW_bf, Hdim, projb, logits, Vdim,
        Bdim, Vdim, Hdim);
    argmax_embed_kernel<<<Bdim, 256, 0, stream>>>(logits, embed, xbuf);
    // attention + state mixing
    havg_kernel<<<cdiv(BH, 256), 256, 0, stream>>>(hcur, havg);
    gemm_f32a_nb<4><<<g_hid, 256, 0, stream>>>(
        havg, Hdim, ahW_bf, Hdim, ahb, hid, Hdim,
        Bdim, Hdim, Hdim);
    attn_kernel<<<Bdim, 256, 0, stream>>>(enc3, hid, aoW, aob, feat, hatt);
    lin_kernel<<<g_lin, 256, 0, stream>>>(hcur, hatt, linW_bf, linb, hnext);
    float* tmp = hcur; hcur = hnext; hnext = tmp;
  }

  // ---- final projection: [B,V,T] output, coalesced along t ----
  {
    const int tiles = ((Vdim >> 4) / 5) * (T_PAD >> 4);              // 375 per batch row
    dim3 grid(cdiv(tiles, WPB), Bdim);
    final_proj_kernel<<<grid, 256, 0, stream>>>(projW_bf, pred_bf, projb, out);
  }
}
